// Block_57492432224435
// MI455X (gfx1250) — matrix operand order repaired and checked by reading, not compiled, not run on hardware
//
#include <hip/hip_runtime.h>
#include <hip/hip_bf16.h>

// ---------------- problem constants ----------------
#define Bc 4
#define Lc 1024
#define Dc 1024
#define Hc 16
#define Fc 4096
#define Ec 4
#define HDc 64
#define NT (Bc * Lc)   // 4096 token rows

// ---------------- types ----------------
typedef __attribute__((ext_vector_type(16))) __bf16 v16bf;
typedef __attribute__((ext_vector_type(8)))  __bf16 v8bf;
typedef __attribute__((ext_vector_type(8)))  float  v8f;
typedef __attribute__((ext_vector_type(4)))  unsigned int v4u;
typedef __attribute__((ext_vector_type(8)))  int v8i;
typedef __attribute__((ext_vector_type(4)))  int v4i;

union bfrag_u { v16bf v; v8bf h[2]; };

static __device__ inline __bf16 f2bf(float f) {
  union { float f; unsigned u; } a; a.f = f;
  unsigned r = a.u + 0x7FFFu + ((a.u >> 16) & 1u);   // round-to-nearest-even
  union { unsigned short s; __bf16 b; } o; o.s = (unsigned short)(r >> 16);
  return o.b;
}
static __device__ inline float bf2f(__bf16 b) {
  union { __bf16 b; unsigned short s; } i; i.b = b;
  union { unsigned u; float f; } o; o.u = ((unsigned)i.s) << 16;
  return o.f;
}

// ---------------- elementwise kernels ----------------
__global__ void k_f32_to_bf16(const float* __restrict__ in, __bf16* __restrict__ out,
                              long long n) {
  long long i = blockIdx.x * 256LL + threadIdx.x;
  if (i < n) out[i] = f2bf(in[i]);
}

__global__ void k_add(const float* __restrict__ a, const float* __restrict__ b,
                      float* __restrict__ out, long long n) {
  long long i = blockIdx.x * 256LL + threadIdx.x;
  if (i < n) out[i] = a[i] + b[i];
}

__global__ void k_silu_mul(const float* __restrict__ h1, const float* __restrict__ h2,
                           __bf16* __restrict__ t, long long n) {
  long long i = blockIdx.x * 256LL + threadIdx.x;
  if (i < n) {
    float x = h1[i];
    float s = x / (1.0f + __expf(-x));
    t[i] = f2bf(s * h2[i]);
  }
}

__global__ void k_accum(float* __restrict__ out, const float* __restrict__ ye,
                        const float* __restrict__ wc, int e, long long n) {
  long long i = blockIdx.x * 256LL + threadIdx.x;
  if (i < n) {
    int row = (int)(i / Dc);
    out[i] += wc[row * Ec + e] * ye[i];
  }
}

// rotary on (B,L,H,HD) fp32 in place; one thread per even/odd pair
__global__ void k_rotary(float* __restrict__ x) {
  long long idx = blockIdx.x * 256LL + threadIdx.x;
  const long long npairs = (long long)Bc * Lc * Hc * (HDc / 2);
  if (idx >= npairs) return;
  int i = (int)(idx % (HDc / 2));
  long long rem = idx / (HDc / 2);      // (b*L + l)*H + h
  long long bl = rem / Hc;
  int pos = (int)(bl % Lc);
  float inv = __powf(10000.0f, (2.0f * (float)i) / (float)HDc);
  float t = (float)pos / inv;
  float s, c;
  __sincosf(t, &s, &c);
  float* p = x + rem * HDc + 2 * i;
  float x0 = p[0], x1 = p[1];
  p[0] = x0 * c - x1 * s;
  p[1] = x0 * s + x1 * c;
}

// v (B,L,H*HD) f32  ->  vT (B,H,HD,L) bf16
__global__ void k_vT(const float* __restrict__ v, __bf16* __restrict__ vT) {
  long long idx = blockIdx.x * 256LL + threadIdx.x;
  const long long n = (long long)Bc * Hc * HDc * Lc;
  if (idx >= n) return;
  int l = (int)(idx % Lc);
  long long r = idx / Lc;
  int hd = (int)(r % HDc);
  long long r2 = r / HDc;
  int h = (int)(r2 % Hc);
  int b = (int)(r2 / Hc);
  vT[idx] = f2bf(v[((long long)(b * Lc + l)) * Dc + h * HDc + hd]);
}

// ---------------- rmsnorm (block per row) ----------------
__global__ __launch_bounds__(256)
void k_rmsnorm_bf16(const float* __restrict__ x, const float* __restrict__ w,
                    __bf16* __restrict__ out) {
  __shared__ float red[256];
  int row = blockIdx.x;
  const float* xr = x + (long long)row * Dc;
  float s = 0.0f;
  for (int d = threadIdx.x; d < Dc; d += 256) { float v = xr[d]; s += v * v; }
  red[threadIdx.x] = s;
  __syncthreads();
  for (int st = 128; st > 0; st >>= 1) {
    if ((int)threadIdx.x < st) red[threadIdx.x] += red[threadIdx.x + st];
    __syncthreads();
  }
  float r = rsqrtf(red[0] / (float)Dc + 1e-6f);
  for (int d = threadIdx.x; d < Dc; d += 256)
    out[(long long)row * Dc + d] = f2bf(w[d] * xr[d] * r);
}

// ---------------- WMMA GEMM: C[M,N] = A[M,K] * B[N,K]^T (bf16 in, f32 out) ----
// 256 threads = 8 waves. Block tile: 128 rows x 64 cols. Wave w owns rows
// [m0+16w, m0+16w+16) and all 64 cols (4 WMMA N-tiles reusing one A fragment).
// The 64x32 B tile is DMA'd into LDS by the Tensor Data Mover (wave 0 issues a
// 2D descriptor), double-buffered and synchronized with s_wait_tensorcnt.
__global__ __launch_bounds__(256)
void k_gemm_bf16(const __bf16* __restrict__ A, const __bf16* __restrict__ Bm,
                 float* __restrict__ C, int M, int N, int K) {
  __shared__ __attribute__((aligned(128))) __bf16 Bs[2][64 * 32];  // [buf][n][k]
  const int lane = threadIdx.x & 31;
  const int wave = threadIdx.x >> 5;
  const int half = lane >> 4;
  const int l15  = lane & 15;
  const int m0 = blockIdx.x * 128 + wave * 16;
  const int n0 = blockIdx.y * 64;

  // ---- TDM descriptor group 1 (constant): data_size=2B, tile 32(k) x 64(n),
  //      tensor_dim0=32, tensor_dim1=64, dim0 stride = K elements.
  v8i g1;
  g1[0] = 0x00010000;                 // workgroup_mask=0, data_size=1 (2 bytes)
  g1[1] = (32 << 16);                 // tensor_dim0[15:0] << 16
  g1[2] = (64 << 16);                 // tensor_dim0 hi=0 | tensor_dim1[15:0]<<16
  g1[3] = (32 << 16);                 // tensor_dim1 hi=0 | tile_dim0=32
  g1[4] = 64;                         // tile_dim1=64, tile_dim2=0
  g1[5] = K;                          // tensor_dim0_stride low 32
  g1[6] = 0;
  g1[7] = 0;
  const v4i gz4 = {0, 0, 0, 0};
  const v8i gz8 = {0, 0, 0, 0, 0, 0, 0, 0};

  const unsigned lds_off[2] = {
      (unsigned)(unsigned long long)(const void*)&Bs[0][0],
      (unsigned)(unsigned long long)(const void*)&Bs[1][0]};
  const unsigned long long gbase =
      (unsigned long long)(const void*)(Bm + (long long)n0 * K);

  auto tdm_issue = [&](int kb, int buf) {
    unsigned long long ga = gbase + (unsigned long long)kb * 2u;
    v4u g0;
    g0[0] = 1u;                                        // count=1 valid descriptor
    g0[1] = lds_off[buf];                              // lds_addr
    g0[2] = (unsigned)(ga & 0xFFFFFFFFu);              // global_addr[31:0]
    g0[3] = (unsigned)((ga >> 32) & 0x01FFFFFFu) | (2u << 30);  // addr hi | type=2
    __builtin_amdgcn_tensor_load_to_lds(g0, g1, gz4, gz4, gz8, 0);
  };

  if (threadIdx.x < 32) tdm_issue(0, 0);               // prologue: tile 0 -> buf 0

  v8f acc[4] = {v8f{}, v8f{}, v8f{}, v8f{}};
  int cur = 0;
  for (int kb = 0; kb < K; kb += 32) {
    if (threadIdx.x < 32) {
      if (kb + 32 < K) {
        tdm_issue(kb + 32, cur ^ 1);                   // prefetch next tile
        __builtin_amdgcn_s_wait_tensorcnt((short)1);   // current tile landed
      } else {
        __builtin_amdgcn_s_wait_tensorcnt((short)0);
      }
    }
    __syncthreads();                                   // buffer `cur` visible to all

    // A fragment (16x32, ISA A-layout): lanes 0-15 row=l15 K{0..7,16..23};
    // lanes 16-31 row=l15 K{8..15,24..31}
    const __bf16* arow = A + (long long)(m0 + l15) * K + kb + half * 8;
    bfrag_u au;
    au.h[0] = *(const v8bf*)(arow);
    au.h[1] = *(const v8bf*)(arow + 16);

#pragma unroll
    for (int t = 0; t < 4; ++t) {
      // B fragment (32x16): lanes 0-15 col N=l15 K{0..7,16..23}; lanes 16-31 K{8..15,24..31}
      const __bf16* brow = &Bs[cur][(t * 16 + l15) * 32 + half * 8];
      bfrag_u bu;
      bu.h[0] = *(const v8bf*)(brow);
      bu.h[1] = *(const v8bf*)(brow + 16);
      acc[t] = __builtin_amdgcn_wmma_f32_16x16x32_bf16(false, au.v, false, bu.v,
                                                       (short)0, acc[t], false, false);
    }
    __syncthreads();   // all waves done reading `cur` before TDM reuses it
    cur ^= 1;
  }

  // C fragment layout: VGPR i -> M = i + 8*half, N = l15
  const int cm = m0 + half * 8;
#pragma unroll
  for (int t = 0; t < 4; ++t) {
    const int cn = n0 + t * 16 + l15;
#pragma unroll
    for (int i = 0; i < 8; ++i)
      C[(long long)(cm + i) * N + cn] = acc[t][i];
  }
}

// ---------------- flash attention (1 wave per 16-row q tile per (b,h)) ------
// qb,kb: (B,L,D) bf16 ; vT: (B,H,HD,L) bf16 ; o: (B,L,D) f32
__global__ __launch_bounds__(32)
void k_flash_attn(const __bf16* __restrict__ qb, const __bf16* __restrict__ kb,
                  const __bf16* __restrict__ vT, float* __restrict__ o) {
  const int lane = threadIdx.x;
  const int half = lane >> 4;
  const int l15  = lane & 15;
  const int q0 = blockIdx.x * 16;
  const int h  = blockIdx.y;
  const int b  = blockIdx.z;

  __shared__ __attribute__((aligned(16))) __bf16 Ps[16 * 32];

  // Q fragments for d in [0,32) and [32,64)
  v16bf qfrag[2];
  {
    const __bf16* base = qb + ((long long)(b * Lc + q0 + l15)) * Dc + h * HDc;
#pragma unroll
    for (int c = 0; c < 2; ++c) {
      bfrag_u u;
      u.h[0] = *(const v8bf*)(base + c * 32 + half * 8);
      u.h[1] = *(const v8bf*)(base + c * 32 + half * 8 + 16);
      qfrag[c] = u.v;
    }
  }

  v8f acc[4] = {v8f{}, v8f{}, v8f{}, v8f{}};
  float m_run[8], l_run[8];
#pragma unroll
  for (int i = 0; i < 8; ++i) { m_run[i] = -1e30f; l_run[i] = 0.0f; }

  const float scale = 0.125f;           // 1/sqrt(HD)
  const int jmax = (q0 + 15) / 32;
  for (int j = 0; j <= jmax; ++j) {
    const int key0 = j * 32;
    v8f S0 = {}, S1 = {};
#pragma unroll
    for (int c = 0; c < 2; ++c) {
      // B fragment of K^T: column = key, d in two 8-runs (table order) per lane
      const __bf16* kbase0 =
          kb + ((long long)(b * Lc + key0 + l15)) * Dc + h * HDc + c * 32 + half * 8;
      bfrag_u u0; u0.h[0] = *(const v8bf*)kbase0; u0.h[1] = *(const v8bf*)(kbase0 + 16);
      S0 = __builtin_amdgcn_wmma_f32_16x16x32_bf16(false, qfrag[c], false, u0.v,
                                                   (short)0, S0, false, false);
      const __bf16* kbase1 = kbase0 + (long long)16 * Dc;
      bfrag_u u1; u1.h[0] = *(const v8bf*)kbase1; u1.h[1] = *(const v8bf*)(kbase1 + 16);
      S1 = __builtin_amdgcn_wmma_f32_16x16x32_bf16(false, qfrag[c], false, u1.v,
                                                   (short)0, S1, false, false);
    }

    // scale + causal mask + online softmax (each row lives in one 16-lane group)
    float p0[8], p1[8];
#pragma unroll
    for (int i = 0; i < 8; ++i) {
      const int m_g = q0 + half * 8 + i;
      float s0 = S0[i] * scale;
      float s1 = S1[i] * scale;
      if (key0 + l15 > m_g)      s0 = -1e30f;
      if (key0 + 16 + l15 > m_g) s1 = -1e30f;
      float mx = fmaxf(s0, s1);
      for (int d = 8; d >= 1; d >>= 1) mx = fmaxf(mx, __shfl_xor(mx, d, 32));
      const float m_new = fmaxf(m_run[i], mx);
      const float alpha = __expf(m_run[i] - m_new);
      const float e0 = __expf(s0 - m_new);
      const float e1 = __expf(s1 - m_new);
      float rs = e0 + e1;
      for (int d = 8; d >= 1; d >>= 1) rs += __shfl_xor(rs, d, 32);
      l_run[i] = l_run[i] * alpha + rs;
      m_run[i] = m_new;
      p0[i] = e0; p1[i] = e1;
#pragma unroll
      for (int t = 0; t < 4; ++t) acc[t][i] *= alpha;
    }

    // re-layout P (C-fragment) -> A-fragment via LDS
#pragma unroll
    for (int i = 0; i < 8; ++i) {
      const int r = half * 8 + i;
      Ps[r * 32 + l15]      = f2bf(p0[i]);
      Ps[r * 32 + 16 + l15] = f2bf(p1[i]);
    }
    __syncthreads();
    bfrag_u pu;
    pu.h[0] = *(const v8bf*)&Ps[l15 * 32 + half * 8];
    pu.h[1] = *(const v8bf*)&Ps[l15 * 32 + half * 8 + 16];

    // P (16x32) x V (32x64) -> 4 accumulator tiles
#pragma unroll
    for (int t = 0; t < 4; ++t) {
      const __bf16* vbase =
          vT + (((long long)(b * Hc + h) * HDc) + t * 16 + l15) * Lc + key0 + half * 8;
      bfrag_u vu; vu.h[0] = *(const v8bf*)vbase; vu.h[1] = *(const v8bf*)(vbase + 16);
      acc[t] = __builtin_amdgcn_wmma_f32_16x16x32_bf16(false, pu.v, false, vu.v,
                                                       (short)0, acc[t], false, false);
    }
    __syncthreads();
  }

  // write O normalized by row sums
#pragma unroll
  for (int t = 0; t < 4; ++t)
#pragma unroll
    for (int i = 0; i < 8; ++i) {
      const int row = q0 + half * 8 + i;
      o[((long long)(b * Lc + row)) * Dc + h * HDc + t * 16 + l15] =
          acc[t][i] / l_run[i];
    }
}

// ---------------- gating: top-2 softmax over E=4 ----------------
__global__ void k_gating(const __bf16* __restrict__ xn, const __bf16* __restrict__ wg,
                         float* __restrict__ wc) {
  int row = blockIdx.x * blockDim.x + threadIdx.x;
  if (row >= NT) return;
  float g[Ec];
  for (int e = 0; e < Ec; ++e) {
    float s = 0.0f;
    const __bf16* x = xn + (long long)row * Dc;
    const __bf16* w = wg + (long long)e * Dc;
    for (int d = 0; d < Dc; ++d) s += bf2f(x[d]) * bf2f(w[d]);
    g[e] = s;
  }
  int i1 = 0;
  for (int e = 1; e < Ec; ++e) if (g[e] > g[i1]) i1 = e;
  int i2 = -1;
  for (int e = 0; e < Ec; ++e) if (e != i1 && (i2 < 0 || g[e] > g[i2])) i2 = e;
  float e1 = 1.0f;                       // exp(g1 - g1)
  float e2 = __expf(g[i2] - g[i1]);
  float inv = 1.0f / (e1 + e2);
  for (int e = 0; e < Ec; ++e) wc[row * Ec + e] = 0.0f;
  wc[row * Ec + i1] = e1 * inv;
  wc[row * Ec + i2] = e2 * inv;
}

// ---------------- host orchestration ----------------
static inline long long cdiv(long long a, long long b) { return (a + b - 1) / b; }

extern "C" void kernel_launch(void* const* d_in, const int* in_sizes, int n_in,
                              void* d_out, int out_size, void* d_ws, size_t ws_size,
                              hipStream_t stream) {
  (void)in_sizes; (void)n_in; (void)out_size; (void)ws_size;
  const float* xmat = (const float*)d_in[0];
  // d_in[1] = mask (causal, unused — masking is analytic)
  const float* wq = (const float*)d_in[2];
  const float* wk = (const float*)d_in[3];
  const float* wv = (const float*)d_in[4];
  const float* wo = (const float*)d_in[5];
  const float* wg = (const float*)d_in[6];
  const float* W1 = (const float*)d_in[7];
  const float* W2 = (const float*)d_in[8];
  const float* W3 = (const float*)d_in[9];
  const float* n1w = (const float*)d_in[10];
  const float* n2w = (const float*)d_in[11];
  float* out = (float*)d_out;

  // workspace carve-up
  char* p = (char*)d_ws;
  auto alloc = [&](size_t bytes) { char* r = p; p += (bytes + 255) & ~(size_t)255; return r; };
  const size_t DD = (size_t)Dc * Dc;
  __bf16* wq_b = (__bf16*)alloc(DD * 2);
  __bf16* wk_b = (__bf16*)alloc(DD * 2);
  __bf16* wv_b = (__bf16*)alloc(DD * 2);
  __bf16* wo_b = (__bf16*)alloc(DD * 2);
  __bf16* wg_b = (__bf16*)alloc((size_t)Ec * Dc * 2);
  __bf16* W1_b = (__bf16*)alloc((size_t)Ec * Fc * Dc * 2);
  __bf16* W2_b = (__bf16*)alloc((size_t)Ec * Fc * Dc * 2);
  __bf16* W3_b = (__bf16*)alloc((size_t)Ec * Dc * Fc * 2);
  __bf16* xn1_b = (__bf16*)alloc((size_t)NT * Dc * 2);
  __bf16* xn2_b = (__bf16*)alloc((size_t)NT * Dc * 2);
  float* q_f = (float*)alloc((size_t)NT * Dc * 4);
  float* k_f = (float*)alloc((size_t)NT * Dc * 4);
  float* v_f = (float*)alloc((size_t)NT * Dc * 4);
  __bf16* qb = (__bf16*)alloc((size_t)NT * Dc * 2);
  __bf16* kbb = (__bf16*)alloc((size_t)NT * Dc * 2);
  __bf16* vT_b = (__bf16*)alloc((size_t)NT * Dc * 2);
  float* o_f = (float*)alloc((size_t)NT * Dc * 4);
  __bf16* ob = (__bf16*)alloc((size_t)NT * Dc * 2);
  float* attnp_f = (float*)alloc((size_t)NT * Dc * 4);
  float* h_f = (float*)alloc((size_t)NT * Dc * 4);
  float* wcomb = (float*)alloc((size_t)NT * Ec * 4);
  float* h1_f = (float*)alloc((size_t)NT * Fc * 4);
  float* h2_f = (float*)alloc((size_t)NT * Fc * 4);
  __bf16* t_b = (__bf16*)alloc((size_t)NT * Fc * 2);
  float* ye_f = (float*)alloc((size_t)NT * Dc * 4);

  auto conv = [&](const float* src, __bf16* dst, long long n) {
    k_f32_to_bf16<<<dim3((unsigned)cdiv(n, 256)), 256, 0, stream>>>(src, dst, n);
  };
  auto gemm = [&](const __bf16* A, const __bf16* Bm, float* Cm, int M, int N, int K) {
    dim3 grid(M / 128, N / 64);
    k_gemm_bf16<<<grid, 256, 0, stream>>>(A, Bm, Cm, M, N, K);
  };

  // weights -> bf16
  conv(wq, wq_b, (long long)DD);
  conv(wk, wk_b, (long long)DD);
  conv(wv, wv_b, (long long)DD);
  conv(wo, wo_b, (long long)DD);
  conv(wg, wg_b, (long long)Ec * Dc);
  conv(W1, W1_b, (long long)Ec * Fc * Dc);
  conv(W2, W2_b, (long long)Ec * Fc * Dc);
  conv(W3, W3_b, (long long)Ec * Dc * Fc);

  // ---- attention ----
  k_rmsnorm_bf16<<<NT, 256, 0, stream>>>(xmat, n1w, xn1_b);
  gemm(xn1_b, wq_b, q_f, NT, Dc, Dc);
  gemm(xn1_b, wk_b, k_f, NT, Dc, Dc);
  gemm(xn1_b, wv_b, v_f, NT, Dc, Dc);

  const long long npairs = (long long)Bc * Lc * Hc * (HDc / 2);
  k_rotary<<<dim3((unsigned)cdiv(npairs, 256)), 256, 0, stream>>>(q_f);
  k_rotary<<<dim3((unsigned)cdiv(npairs, 256)), 256, 0, stream>>>(k_f);

  conv(q_f, qb, (long long)NT * Dc);
  conv(k_f, kbb, (long long)NT * Dc);
  k_vT<<<dim3((unsigned)cdiv((long long)NT * Dc, 256)), 256, 0, stream>>>(v_f, vT_b);

  dim3 fgrid(Lc / 16, Hc, Bc);
  k_flash_attn<<<fgrid, 32, 0, stream>>>(qb, kbb, vT_b, o_f);

  conv(o_f, ob, (long long)NT * Dc);
  gemm(ob, wo_b, attnp_f, NT, Dc, Dc);

  const long long nTD = (long long)NT * Dc;
  k_add<<<dim3((unsigned)cdiv(nTD, 256)), 256, 0, stream>>>(xmat, attnp_f, h_f, nTD);

  // ---- MoE ----
  k_rmsnorm_bf16<<<NT, 256, 0, stream>>>(h_f, n2w, xn2_b);
  k_gating<<<dim3((NT + 127) / 128), 128, 0, stream>>>(xn2_b, wg_b, wcomb);

  (void)hipMemcpyAsync(d_out, h_f, nTD * sizeof(float), hipMemcpyDeviceToDevice, stream);

  const long long nTF = (long long)NT * Fc;
  for (int e = 0; e < Ec; ++e) {
    gemm(xn2_b, W1_b + (size_t)e * Fc * Dc, h1_f, NT, Fc, Dc);
    gemm(xn2_b, W2_b + (size_t)e * Fc * Dc, h2_f, NT, Fc, Dc);
    k_silu_mul<<<dim3((unsigned)cdiv(nTF, 256)), 256, 0, stream>>>(h1_f, h2_f, t_b, nTF);
    gemm(t_b, W3_b + (size_t)e * Dc * Fc, ye_f, NT, Dc, Fc);
    k_accum<<<dim3((unsigned)cdiv(nTD, 256)), 256, 0, stream>>>(out, ye_f, wcomb, e, nTD);
  }
}
